// DoRALinear_55087250538729
// MI455X (gfx1250) — matrix operand order repaired and checked by reading, not compiled, not run on hardware
//
#include <hip/hip_runtime.h>
#include <stdint.h>

typedef __bf16 bf16_t;
typedef __attribute__((ext_vector_type(16))) __bf16 v16bf;
typedef __attribute__((ext_vector_type(8)))  __bf16 v8bf;
typedef __attribute__((ext_vector_type(8)))  float  v8f;
typedef __attribute__((ext_vector_type(4)))  float  v4f;

#define IN_F   4096
#define OUT_F  4096
#define RANK   16
#define BM     128
#define BN     128
#define BK     64
#define LDS_K  72   // 64 + 8 pad -> 144B row stride (16B aligned, 36-bank step)

__device__ __forceinline__ bf16_t f2bf(float f) {
  union { float f; uint32_t u; } v; v.f = f;
  uint32_t r = v.u + 0x7FFFu + ((v.u >> 16) & 1u);   // round-to-nearest-even
  uint16_t h = (uint16_t)(r >> 16);
  bf16_t b;
  __builtin_memcpy(&b, &h, sizeof(h));
  return b;
}

union Frag {
  v16bf  v;
  v8bf   h[2];
  bf16_t e[16];
};

// ---- CDNA5 async copy: memory -> LDS, tracked by ASYNCcnt (inline asm path) ----
__device__ __forceinline__ void async_load_b128(const bf16_t* g, const bf16_t* l) {
  // generic shared-aperture address: low 32 bits == LDS byte offset (ISA 10.2)
  uint32_t laddr = (uint32_t)(uintptr_t)l;
  uint64_t gaddr = (uint64_t)(uintptr_t)g;
  asm volatile("global_load_async_to_lds_b128 %0, %1, off"
               :: "v"(laddr), "v"(gaddr) : "memory");
}
template <int N>
__device__ __forceinline__ void wait_asynccnt() {
  asm volatile("s_wait_asynccnt %0" :: "n"(N) : "memory");
}

// ---------------------------------------------------------------------------
// Kernel 1: scale[r] = mag[r]/||W[r,:]+B[r,:]@A||; Bb[r,:] = bf16(B[r,:]*scale)
// ---------------------------------------------------------------------------
__global__ __launch_bounds__(256)
void dora_scale_kernel(const float* __restrict__ W,
                       const float* __restrict__ A,
                       const float* __restrict__ B,
                       const float* __restrict__ mag,
                       float* __restrict__ scale,
                       bf16_t* __restrict__ Bb) {
  const int row = blockIdx.x;
  const int tid = threadIdx.x;
  __shared__ float sB[RANK];
  __shared__ float red[256];
  __shared__ float s_scale;
  if (tid < RANK) sB[tid] = B[(size_t)row * RANK + tid];
  __syncthreads();
  float acc = 0.f;
  for (int c = tid; c < IN_F; c += 256) {
    float l = 0.f;
#pragma unroll
    for (int r = 0; r < RANK; ++r) l += sB[r] * A[(size_t)r * IN_F + c];
    float v = W[(size_t)row * IN_F + c] + l;
    acc += v * v;
  }
  red[tid] = acc;
  __syncthreads();
  for (int s = 128; s > 0; s >>= 1) {
    if (tid < s) red[tid] += red[tid + s];
    __syncthreads();
  }
  if (tid == 0) {
    float s = mag[row] * rsqrtf(red[0]);
    scale[row] = s;
    s_scale = s;
  }
  __syncthreads();
  if (tid < RANK) Bb[(size_t)row * RANK + tid] = f2bf(sB[tid] * s_scale);
}

// ---------------------------------------------------------------------------
// Kernel 2: Xb = bf16(x)          (one block per token row, 16 cols/thread)
// ---------------------------------------------------------------------------
__global__ __launch_bounds__(256)
void cvt_x_kernel(const float* __restrict__ X, bf16_t* __restrict__ Xb) {
  const int row = blockIdx.x;
  const int c0  = threadIdx.x * 16;
  const float* p = X + (size_t)row * IN_F + c0;
  Frag u;
#pragma unroll
  for (int i = 0; i < 4; ++i) {
    v4f v = *(const v4f*)(p + i * 4);
#pragma unroll
    for (int j = 0; j < 4; ++j) u.e[i * 4 + j] = f2bf(v[j]);
  }
  bf16_t* q = Xb + (size_t)row * IN_F + c0;
  *(v8bf*)q       = u.h[0];
  *(v8bf*)(q + 8) = u.h[1];
}

// ---------------------------------------------------------------------------
// Kernel 3: Wb[o,:] = bf16(W[o,:] * scale[o])   (magnitude folded into W)
// ---------------------------------------------------------------------------
__global__ __launch_bounds__(256)
void cvt_w_kernel(const float* __restrict__ W, const float* __restrict__ scale,
                  bf16_t* __restrict__ Wb) {
  const int row = blockIdx.x;
  const int c0  = threadIdx.x * 16;
  const float s = scale[row];
  const float* p = W + (size_t)row * IN_F + c0;
  Frag u;
#pragma unroll
  for (int i = 0; i < 4; ++i) {
    v4f v = *(const v4f*)(p + i * 4);
#pragma unroll
    for (int j = 0; j < 4; ++j) u.e[i * 4 + j] = f2bf(v[j] * s);
  }
  bf16_t* q = Wb + (size_t)row * IN_F + c0;
  *(v8bf*)q       = u.h[0];
  *(v8bf*)(q + 8) = u.h[1];
}

// ---------------------------------------------------------------------------
// Kernel 4: xab[tok,r] = bf16(x[tok,:] . A[r,:])
// ---------------------------------------------------------------------------
__global__ __launch_bounds__(256)
void dora_xa_kernel(const float* __restrict__ x,
                    const float* __restrict__ A,
                    bf16_t* __restrict__ xab) {
  const int tok  = blockIdx.x;
  const int tid  = threadIdx.x;
  const int rank = tid & 15;
  const int chnk = tid >> 4;
  __shared__ float red[256];
  float acc = 0.f;
  for (int c = chnk; c < IN_F; c += 16)
    acc += x[(size_t)tok * IN_F + c] * A[(size_t)rank * IN_F + c];
  red[tid] = acc;
  __syncthreads();
  if (tid < RANK) {
    float s = 0.f;
#pragma unroll
    for (int j = 0; j < 16; ++j) s += red[tid + 16 * j];
    xab[(size_t)tok * RANK + tid] = f2bf(s);
  }
}

// ---------------------------------------------------------------------------
// Kernel 5: out = Xb @ Wb^T + xab @ Bb^T    (scale pre-folded into Wb, Bb)
// 128x128 tile, BK=64, async-to-LDS double buffered, 16 WMMA/iter/wave.
// ---------------------------------------------------------------------------
__global__ __launch_bounds__(256)
void dora_gemm_kernel(const bf16_t* __restrict__ Xb,   // [NTOK, IN_F]
                      const bf16_t* __restrict__ Wb,   // [OUT_F, IN_F]
                      const bf16_t* __restrict__ xab,  // [NTOK, RANK]
                      const bf16_t* __restrict__ Bb,   // [OUT_F, RANK]
                      float* __restrict__ out) {       // [NTOK, OUT_F]
  __shared__ __align__(16) bf16_t sX[2][BM][LDS_K];    // 2 x 18KB
  __shared__ __align__(16) bf16_t sW[2][BN][LDS_K];    // 2 x 18KB

  const int tid    = threadIdx.x;
  const int wid    = tid >> 5;
  const int lane   = tid & 31;
  const int lrow   = lane & 15;
  const int lhalf  = lane >> 4;
  const int m_tile = blockIdx.y * BM;
  const int n_tile = blockIdx.x * BN;
  const int wm     = (wid & 3) * 32;
  const int wn     = (wid >> 2) * 64;

  v8f acc[2][4];
#pragma unroll
  for (int mi = 0; mi < 2; ++mi)
#pragma unroll
    for (int ni = 0; ni < 4; ++ni) acc[mi][ni] = (v8f){};

  // 1024 16B-chunks per tile per iter; 4 per thread per tile
  auto issue = [&](int k0, int buf) {
#pragma unroll
    for (int i = 0; i < 4; ++i) {
      const int chunk = tid + 256 * i;
      const int row = chunk >> 3;
      const int seg = (chunk & 7) * 8;   // 8 bf16 = 16B
      async_load_b128(Xb + (size_t)(m_tile + row) * IN_F + k0 + seg,
                      &sX[buf][row][seg]);
      async_load_b128(Wb + (size_t)(n_tile + row) * IN_F + k0 + seg,
                      &sW[buf][row][seg]);
    }
  };

  const int NIT = IN_F / BK;   // 64
  issue(0, 0);
  for (int it = 0; it < NIT; ++it) {
    const int buf = it & 1;
    if (it + 1 < NIT) {
      issue((it + 1) * BK, buf ^ 1);   // prefetch next tile into other buffer
      wait_asynccnt<8>();              // this iter's 8 loads done (in-order)
    } else {
      wait_asynccnt<0>();
    }
    __syncthreads();

#pragma unroll
    for (int ks = 0; ks < 2; ++ks) {   // two K=32 steps within BK=64
      Frag a[2], b[4];
#pragma unroll
      for (int mi = 0; mi < 2; ++mi) {
        const bf16_t* p = &sX[buf][wm + mi * 16 + lrow][ks * 32 + lhalf * 8];
        a[mi].h[0] = *(const v8bf*)p;         // K 0-7 / 8-15
        a[mi].h[1] = *(const v8bf*)(p + 16);  // K 16-23 / 24-31
      }
#pragma unroll
      for (int ni = 0; ni < 4; ++ni) {
        const bf16_t* p = &sW[buf][wn + ni * 16 + lrow][ks * 32 + lhalf * 8];
        b[ni].h[0] = *(const v8bf*)p;         // K 0-7 / 8-15
        b[ni].h[1] = *(const v8bf*)(p + 16);  // K 16-23 / 24-31
      }
#pragma unroll
      for (int mi = 0; mi < 2; ++mi)
#pragma unroll
        for (int ni = 0; ni < 4; ++ni)
          acc[mi][ni] = __builtin_amdgcn_wmma_f32_16x16x32_bf16(
              false, a[mi].v, false, b[ni].v, (short)0, acc[mi][ni], false, false);
    }
    __syncthreads();
  }

  // ---- fused LoRA: one extra WMMA per accumulator (rank 16 padded to K=32) --
  const bf16_t bz = f2bf(0.0f);
  Frag xaf[2], lbf[4];
#pragma unroll
  for (int mi = 0; mi < 2; ++mi) {
    const bf16_t* p = xab + (size_t)(m_tile + wm + mi * 16 + lrow) * RANK + lhalf * 8;
    xaf[mi].h[0] = *(const v8bf*)p;           // K 0-7 / 8-15 (rank halves)
#pragma unroll
    for (int j = 8; j < 16; ++j) xaf[mi].e[j] = bz;   // K >= 16 pad
  }
#pragma unroll
  for (int ni = 0; ni < 4; ++ni) {
    const int o = n_tile + wn + ni * 16 + lrow;
    {                                         // K 0-7 / 8-15 (rank halves)
      const bf16_t* p = Bb + (size_t)o * RANK + lhalf * 8;
      lbf[ni].h[0] = *(const v8bf*)p;
#pragma unroll
      for (int j = 8; j < 16; ++j) lbf[ni].e[j] = bz;
    }
  }
#pragma unroll
  for (int mi = 0; mi < 2; ++mi)
#pragma unroll
    for (int ni = 0; ni < 4; ++ni)
      acc[mi][ni] = __builtin_amdgcn_wmma_f32_16x16x32_bf16(
          false, xaf[mi].v, false, lbf[ni].v, (short)0, acc[mi][ni], false, false);

  // ---- store (scale already folded): C layout VGPR e -> M=e|8+e, N=lane%16 --
#pragma unroll
  for (int ni = 0; ni < 4; ++ni) {
    const int gn = n_tile + wn + ni * 16 + lrow;
#pragma unroll
    for (int mi = 0; mi < 2; ++mi) {
#pragma unroll
      for (int e = 0; e < 8; ++e) {
        const int gm = m_tile + wm + mi * 16 + lhalf * 8 + e;
        out[(size_t)gm * OUT_F + gn] = acc[mi][ni][e];
      }
    }
  }
}

// ---------------------------------------------------------------------------
extern "C" void kernel_launch(void* const* d_in, const int* in_sizes, int n_in,
                              void* d_out, int out_size, void* d_ws, size_t ws_size,
                              hipStream_t stream) {
  const float* x   = (const float*)d_in[0];   // [4,2048,4096]
  const float* W   = (const float*)d_in[1];   // [4096,4096]
  const float* A   = (const float*)d_in[2];   // [16,4096]
  const float* B   = (const float*)d_in[3];   // [4096,16]
  const float* mag = (const float*)d_in[4];   // [4096]
  float* out = (float*)d_out;

  const int ntok = in_sizes[0] / IN_F;        // 8192

  char* w = (char*)d_ws;
  float*  scale = (float*)w;   w += (size_t)OUT_F * sizeof(float);      // 16 KB
  bf16_t* xab   = (bf16_t*)w;  w += (size_t)ntok * RANK * sizeof(bf16_t);   // 256 KB
  bf16_t* Bb    = (bf16_t*)w;  w += (size_t)OUT_F * RANK * sizeof(bf16_t);  // 128 KB
  bf16_t* Xbf   = (bf16_t*)w;  w += (size_t)ntok * IN_F * sizeof(bf16_t);   // 64 MB
  bf16_t* Wbf   = (bf16_t*)w;                                               // 32 MB

  dora_scale_kernel<<<OUT_F, 256, 0, stream>>>(W, A, B, mag, scale, Bb);
  cvt_w_kernel<<<OUT_F, 256, 0, stream>>>(W, scale, Wbf);
  cvt_x_kernel<<<ntok, 256, 0, stream>>>(x, Xbf);
  dora_xa_kernel<<<ntok, 256, 0, stream>>>(x, A, xab);

  dim3 grid(OUT_F / BN, ntok / BM);           // (32, 64)
  dora_gemm_kernel<<<grid, 256, 0, stream>>>(Xbf, Wbf, xab, Bb, out);
}
